// Clahe_27779848470822
// MI455X (gfx1250) — matrix operand order repaired and checked by reading, not compiled, not run on hardware
//
#include <hip/hip_runtime.h>

typedef __attribute__((ext_vector_type(16))) _Float16 v16h;
typedef __attribute__((ext_vector_type(8)))  float    v8f;
typedef __attribute__((ext_vector_type(2)))  float    v2f;
typedef __attribute__((ext_vector_type(4)))  unsigned u32x4;
typedef __attribute__((ext_vector_type(8)))  unsigned u32x8;

#define NB 256
#define CLIP_MAX 640.0f   // max(40.0*4096//256, 1.0)

// ---------------------------------------------------------------------------
// Kernel 1: one block per tile (b,ty,tx,c). 256 threads.
//   - LDS histogram (uint atomics), clip at 640, exact excess redistribution
//   - 256-bin inclusive cumsum fully on the matrix pipe (wave 0):
//       P   = Hm(16x16, K-padded to 32) x U(upper-tri ones)   [f16 WMMA, exact]
//       Off = L_strict x T, T[k][*] = chunk totals            [4x f32 WMMA, exact]
//     LUT = floor(clip((P + Off)*255/4096, 0, 255)) -- bit-exact vs reference.
// ---------------------------------------------------------------------------
__global__ __launch_bounds__(256) void clahe_hist_lut(const float* __restrict__ img,
                                                      float* __restrict__ luts) {
    __shared__ unsigned histU[NB];
    __shared__ float    hf[NB];
    __shared__ float    red[NB];
    __shared__ float    P[NB];
    __shared__ float    off[16];
    __shared__ int      resid_s, redist_s;

    const int tid = threadIdx.x;
    int bid = blockIdx.x;                 // (((b*8+ty)*8+tx)*3+c)
    const int c  = bid % 3;  bid /= 3;
    const int tx = bid & 7;  bid >>= 3;
    const int ty = bid & 7;
    const int b  = bid >> 3;

    histU[tid] = 0u;
    __syncthreads();

    const size_t plane = (size_t)(b * 3 + c) * (size_t)(512 * 512);
    const size_t tbase = plane + (size_t)(ty * 64) * 512 + (size_t)(tx * 64);

    // 4096 pixels per tile, 4 x float4 per thread, coalesced.
    #pragma unroll
    for (int i = 0; i < 4; ++i) {
        const int p  = i * 256 + tid;
        const int ly = p >> 4;
        const int lx = (p & 15) * 4;
        const float4 v4 = *(const float4*)(img + tbase + (size_t)ly * 512 + lx);
        if (i < 3) {
            const int pn = (i + 1) * 256 + tid;
            __builtin_prefetch(img + tbase + (size_t)(pn >> 4) * 512 + (pn & 15) * 4, 0, 0);
        }
        const float pv[4] = {v4.x, v4.y, v4.z, v4.w};
        #pragma unroll
        for (int j = 0; j < 4; ++j) {
            int bin = (int)floorf(pv[j] * 256.0f);
            bin = bin < 0 ? 0 : (bin > 255 ? 255 : bin);
            atomicAdd(&histU[bin], 1u);
        }
    }
    __syncthreads();

    // Clip at 640 and reduce to get total kept mass (all integers, exact in f32).
    const float h = fminf((float)histU[tid], CLIP_MAX);
    red[tid] = h;
    __syncthreads();
    for (int s = 128; s > 0; s >>= 1) {
        if (tid < s) red[tid] += red[tid + s];
        __syncthreads();
    }
    if (tid == 0) {
        const int e = 4096 - (int)red[0];   // excess, exact integer
        resid_s  = e & 255;                 // excess mod 256
        redist_s = e >> 8;                  // (excess - residual)/256
    }
    __syncthreads();
    hf[tid] = h + (float)redist_s + ((tid < resid_s) ? 1.0f : 0.0f);
    __syncthreads();

    // -------- cumsum on the matrix pipe (wave 0 only, EXEC all-ones) ---------
    if (tid < 32) {
        const int lane = tid;
        const int m  = lane & 15;   // A row / D column index
        const int hi = lane >> 4;   // half-wave selector
        v16h a;   // A: 16x32 f16, A[m][k] = hf[m*16+k] for k<16, else 0
        #pragma unroll
        for (int g = 0; g < 8; ++g) {
            #pragma unroll
            for (int sub = 0; sub < 2; ++sub) {
                const int e = g * 2 + sub;
                if (g < 4) {
                    const int k = hi * 8 + g * 2 + sub;      // ISA 16-bit A layout
                    a[e] = (_Float16)hf[m * 16 + k];
                } else {
                    a[e] = (_Float16)0.0f;                   // K = 16..31 pad
                }
            }
        }
        v16h bb;  // B: 32x16 f16, B[k][n] = (k<=n && k<16) ? 1 : 0
        const int n = m;
        #pragma unroll
        for (int e = 0; e < 16; ++e) {
            const int k = (e < 8) ? (hi * 8 + e) : (16 + hi * 8 + (e - 8)); // ISA 16-bit B layout
            bb[e] = (_Float16)((k <= n) ? 1.0f : 0.0f);      // e>=8 -> k>=16>n -> 0
        }
        v8f cz = {0.f, 0.f, 0.f, 0.f, 0.f, 0.f, 0.f, 0.f};
        v8f d = __builtin_amdgcn_wmma_f32_16x16x32_f16(
            /*neg_a=*/false, a, /*neg_b=*/false, bb,
            /*c_mod=*/(short)0, cz, /*reuse_a=*/false, /*reuse_b=*/false);
        // D layout: lane n=L&15, VGPR r holds row (r + hi*8)
        #pragma unroll
        for (int r2 = 0; r2 < 8; ++r2) {
            P[(r2 + hi * 8) * 16 + n] = d[r2];
        }

        // Exclusive chunk offsets: Off = L_strict(16x16) x T, T[k][n] = t[k].
        // K=16 via 4 chained V_WMMA_F32_16X16X4_F32 (exact f32 integer math).
        v8f dOff = {0.f, 0.f, 0.f, 0.f, 0.f, 0.f, 0.f, 0.f};
        #pragma unroll
        for (int o = 0; o < 16; o += 4) {
            v2f a2, b2;
            #pragma unroll
            for (int r = 0; r < 2; ++r) {
                const int k = o + hi * 2 + r;                // ISA 32-bit A/B layout
                a2[r] = (k < m) ? 1.0f : 0.0f;               // L_strict[m][k]
                b2[r] = P[k * 16 + 15];                      // t[k] (chunk total)
            }
            dOff = __builtin_amdgcn_wmma_f32_16x16x4_f32(
                false, a2, false, b2, (short)0, dOff, false, false);
        }
        if (n == 0) {            // lanes 0 and 16 write the 16 offsets
            #pragma unroll
            for (int r2 = 0; r2 < 8; ++r2) off[r2 + hi * 8] = dOff[r2];
        }
    }
    __syncthreads();

    const float cdf = P[tid] + off[tid >> 4];                       // exact integer <= 4096
    const float lv  = floorf(fminf(fmaxf(cdf * (255.0f / 4096.0f), 0.0f), 255.0f));
    luts[(size_t)blockIdx.x * NB + tid] = lv;
}

// ---------------------------------------------------------------------------
// Kernel 2: one block per tile. Preload the 3x3 LUT neighborhood into LDS with
// the Tensor Data Mover (one 2D-tile descriptor per valid ty-row: 256 elems x
// up-to-3 tx rows, stride 768 elems -> lands contiguously on the dx slots),
// then bilinear-blend per pixel. Clamped-out-of-range slots are never read.
// ---------------------------------------------------------------------------
__device__ __forceinline__ void axis_slots(int pos, int tile, int& d0, int& d1, float& w) {
    const bool inside = (pos >= 32) && (pos < 480);
    const int  k  = (pos - 32) >> 6;                // valid only when inside (then 0..6)
    const int  t0 = inside ? k : (pos < 32 ? 0 : 7);
    const int  t1 = inside ? (k + 1) : t0;
    const int  rr = (pos - 32) - (k << 6);
    w  = inside ? ((float)(63 - rr) / 63.0f) : 1.0f;
    d0 = t0 - tile + 1;
    d1 = t1 - tile + 1;
}

__global__ __launch_bounds__(256) void clahe_apply(const float* __restrict__ img,
                                                   const float* __restrict__ luts,
                                                   float* __restrict__ out) {
    __shared__ float slut[9 * NB];

    const int tid = threadIdx.x;
    int bid = blockIdx.x;
    const int c  = bid % 3;  bid /= 3;
    const int tx = bid & 7;  bid >>= 3;
    const int ty = bid & 7;
    const int b  = bid >> 3;

    // ---- TDM preload of up to 9 neighbor LUTs (wave 0 issues, all wait) ----
    if (tid < 32) {
        const int      txs   = (tx > 0) ? tx - 1 : 0;
        const int      txe   = (tx < 7) ? tx + 1 : 7;
        const unsigned nrows = (unsigned)(txe - txs + 1);
        #pragma unroll
        for (int dy = 0; dy < 3; ++dy) {
            const int typ = ty + dy - 1;
            if (typ < 0 || typ > 7) continue;
            const unsigned slot0    = (unsigned)(dy * 3 + (txs - tx + 1));
            const unsigned lds_base = (unsigned)(size_t)&slut[slot0 * NB];
            const unsigned long long gaddr =
                (unsigned long long)(size_t)(luts +
                    (size_t)((((b * 8 + typ) * 8 + txs) * 3) + c) * NB);

            u32x4 g0;
            g0.x = 1u;                                        // count=1, user mode
            g0.y = lds_base;                                  // LDS byte address
            g0.z = (unsigned)(gaddr & 0xFFFFFFFFull);         // global_addr[31:0]
            g0.w = (unsigned)((gaddr >> 32) & 0x01FFFFFFu)    // global_addr[56:32]
                 | (2u << 30);                                // type = 2 ("image")

            u32x8 g1;
            g1.s0 = (2u << 16);          // data_size = 4 bytes; no mask/pad/iterate
            g1.s1 = (256u << 16);        // tensor_dim0 = 256 elems (bits 79:48)
            g1.s2 = (nrows << 16);       // tensor_dim1 = nrows   (bits 111:80)
            g1.s3 = (256u << 16);        // tile_dim0   = 256     (bits 127:112)
            g1.s4 = nrows;               // tile_dim1   = nrows; tile_dim2 = 0
            g1.s5 = 768u;                // tensor_dim0_stride = 3*256 elems
            g1.s6 = 0u;                  // stride hi / tensor_dim1_stride lo
            g1.s7 = 0u;                  // tensor_dim1_stride hi

            // 2-group form == VADDR2/VADDR3 NULL (tensors up to 2D)
            asm volatile("tensor_load_to_lds %0, %1" :: "s"(g0), "s"(g1) : "memory");
        }
        __builtin_amdgcn_s_wait_tensorcnt(0);
    }
    __syncthreads();

    const size_t plane = (size_t)(b * 3 + c) * (size_t)(512 * 512);

    #pragma unroll
    for (int i = 0; i < 4; ++i) {
        const int p  = i * 256 + tid;
        const int ly = p >> 4;
        const int lx = (p & 15) * 4;
        const int y  = ty * 64 + ly;
        const int x0 = tx * 64 + lx;

        int   dy0, dy1;
        float wy;
        axis_slots(y, ty, dy0, dy1, wy);

        const size_t rowoff = plane + (size_t)y * 512 + x0;
        const float4 v4 = *(const float4*)(img + rowoff);
        if (i < 3) {
            const int pn = (i + 1) * 256 + tid;
            __builtin_prefetch(img + plane + (size_t)(ty * 64 + (pn >> 4)) * 512
                                   + (tx * 64 + (pn & 15) * 4), 0, 0);
        }
        const float pv[4] = {v4.x, v4.y, v4.z, v4.w};
        float o[4];
        #pragma unroll
        for (int j = 0; j < 4; ++j) {
            int   dx0, dx1;
            float wx;
            axis_slots(x0 + j, tx, dx0, dx1, wx);

            int v = (int)floorf(pv[j] * 255.0f);
            v = v < 0 ? 0 : (v > 255 ? 255 : v);

            const float l00 = slut[(dy0 * 3 + dx0) * NB + v];
            const float l01 = slut[(dy0 * 3 + dx1) * NB + v];
            const float l10 = slut[(dy1 * 3 + dx0) * NB + v];
            const float l11 = slut[(dy1 * 3 + dx1) * NB + v];

            const float blend = wy * wx * l00 + wy * (1.0f - wx) * l01
                              + (1.0f - wy) * wx * l10 + (1.0f - wy) * (1.0f - wx) * l11;
            o[j] = blend / 255.0f;
        }
        float4 ov = make_float4(o[0], o[1], o[2], o[3]);
        *(float4*)(out + rowoff) = ov;
    }
}

// ---------------------------------------------------------------------------
extern "C" void kernel_launch(void* const* d_in, const int* in_sizes, int n_in,
                              void* d_out, int out_size, void* d_ws, size_t ws_size,
                              hipStream_t stream) {
    (void)in_sizes; (void)n_in; (void)out_size; (void)ws_size;
    const float* img  = (const float*)d_in[0];
    float*       out  = (float*)d_out;
    float*       luts = (float*)d_ws;     // 6144 * 256 floats = 6 MB scratch

    const int ntiles = 32 * 3 * 8 * 8;    // 6144
    clahe_hist_lut<<<ntiles, 256, 0, stream>>>(img, luts);
    clahe_apply  <<<ntiles, 256, 0, stream>>>(img, luts, out);
}
